// RotationNet_5042291605728
// MI455X (gfx1250) — matrix operand order repaired and checked by reading, not compiled, not run on hardware
//
#include <hip/hip_runtime.h>
#include <hip/hip_bf16.h>

// ---------------------------------------------------------------------------
// Problem constants (match reference)
// ---------------------------------------------------------------------------
#define BB   512
#define CC   2048
#define HH   8
#define WW   8
#define VV   15
#define DD   128
#define DIN  2052
#define K0P  2080          // DIN padded to multiple of 32 for WMMA K-steps
#define FF   (VV * DD)     // 1920
#define NHEAD 64           // 51 head outputs padded to 64
#define LDSKA 40           // A-tile K stride: 80B rows (16B-aligned for async b128,
                           //   20-bank step -> conflict-free fragment reads)
#define LDSKB 34           // B-tile K stride (17-bank step, coprime w/ 64)

typedef float        v8f   __attribute__((ext_vector_type(8)));
typedef __bf16       v16bf __attribute__((ext_vector_type(16)));
typedef unsigned int v8u   __attribute__((ext_vector_type(8)));

__device__ __forceinline__ unsigned short f2bf(float f) {
    unsigned int u = __float_as_uint(f);
    u += 0x7FFFu + ((u >> 16) & 1u);           // round-to-nearest-even
    return (unsigned short)(u >> 16);
}

// ---------------------------------------------------------------------------
// Async global -> LDS staging (gfx1250 GLOBAL_LOAD_ASYNC_TO_LDS, ASYNCcnt)
// LDS byte offset = low 32 bits of the generic pointer (ISA 10.2 LDS aperture).
// ---------------------------------------------------------------------------
__device__ __forceinline__ void async_ld_b128(unsigned short* lds_dst,
                                              const unsigned short* g_src) {
    unsigned int lo = (unsigned int)(unsigned long long)lds_dst;
    asm volatile("global_load_async_to_lds_b128 %0, %1, off"
                 :: "v"(lo), "v"(g_src)
                 : "memory");
}
__device__ __forceinline__ void async_wait0() {
    asm volatile("s_wait_asynccnt 0" ::: "memory");
}

// ---------------------------------------------------------------------------
// Fragment load (per ISA 7.12.2 VGPR layouts) + 16x16x32 bf16 WMMA
//   As: [64][LDSKA] row-major (rows = M)
//   Bs: [NT*16][LDSKB] column-major-in-K (rows = N)
// ---------------------------------------------------------------------------
template <int NT>
__device__ __forceinline__ void frag_mma(const unsigned short* As,
                                         const unsigned short* Bs,
                                         int lane, int wave, v8f* acc) {
    const int m    = lane & 15;
    const int half = lane >> 4;
    const int arow = wave * 16 + m;
    v8u au;
#pragma unroll
    for (int r = 0; r < 8; ++r) {
        const int kb = 2 * r + ((r >= 4) ? 8 : 0) + half * 8;   // A K-pair base
        au[r] = *(const unsigned int*)(As + arow * LDSKA + kb);
    }
    const v16bf af = __builtin_bit_cast(v16bf, au);
#pragma unroll
    for (int t = 0; t < NT; ++t) {
        const int ncol = t * 16 + m;
        v8u bu;
#pragma unroll
        for (int r = 0; r < 8; ++r)
            bu[r] = *(const unsigned int*)(Bs + ncol * LDSKB + 2 * r + ((r >= 4) ? 8 : 0) + half * 8);
        const v16bf bf = __builtin_bit_cast(v16bf, bu);
        acc[t] = __builtin_amdgcn_wmma_f32_16x16x32_bf16(
            false, af, false, bf, (short)0, acc[t], false, false);
    }
}

// ---------------------------------------------------------------------------
// Per-vertex GEMM + bias:  Out[v][512][128] = A[v][512][K](bf16) x W[v][K][128](bf16)
// grid = (BB/64, VV), block = 128 (4 waves); wave w owns rows [w*16, w*16+16)
// ---------------------------------------------------------------------------
__global__ void __launch_bounds__(128)
gemm_bias_kernel(const unsigned short* __restrict__ A,
                 const unsigned short* __restrict__ Wt,
                 const float* __restrict__ bias,
                 float* __restrict__ Out, int K) {
    __shared__ unsigned short As[64 * LDSKA];
    __shared__ unsigned short Bs[128 * LDSKB];

    const int tid  = threadIdx.x;
    const int wave = tid >> 5;
    const int lane = tid & 31;
    const int mtile = blockIdx.x;
    const int v     = blockIdx.y;

    const unsigned short* Ag = A  + (size_t)v * BB * K + (size_t)(mtile * 64) * K;
    const unsigned short* Bg = Wt + (size_t)v * K * DD;
    const float*          bv = bias + (size_t)v * DD;
    float*                Og = Out  + (size_t)v * BB * DD;

    v8f acc[8];
#pragma unroll
    for (int t = 0; t < 8; ++t) acc[t] = v8f{0.f,0.f,0.f,0.f,0.f,0.f,0.f,0.f};

    for (int k0 = 0; k0 < K; k0 += 32) {
        __syncthreads();
        // stage A via async DMA: 64 rows x 32 K (bf16), 16B chunks straight to LDS
#pragma unroll
        for (int i = 0; i < 2; ++i) {
            const int chunk = tid + i * 128;          // 0..255
            const int row   = chunk >> 2;
            const int col   = (chunk & 3) * 8;
            async_ld_b128(&As[row * LDSKA + col], Ag + (size_t)row * K + k0 + col);
        }
        // stage B transposed (manual; transpose not expressible as plain DMA):
        // 32 K-rows x 128 N -> Bs[n][k]
#pragma unroll
        for (int i = 0; i < 4; ++i) {
            const int c  = tid + i * 128;             // 0..511
            const int kr = c >> 4;
            const int cb = (c & 15) * 8;
            const uint4 d = *(const uint4*)(Bg + (size_t)(k0 + kr) * DD + cb);
            const unsigned short* s = (const unsigned short*)&d;
#pragma unroll
            for (int j = 0; j < 8; ++j) Bs[(cb + j) * LDSKB + kr] = s[j];
        }
        if (k0 + 32 < K) {                            // hint the next K-slab
            __builtin_prefetch(Ag + (size_t)(tid >> 1) * K + k0 + 32, 0, 1);
            __builtin_prefetch(Bg + (size_t)(k0 + 32 + (tid & 31)) * DD, 0, 1);
        }
        async_wait0();                                // our DMA landed in LDS
        __syncthreads();
        frag_mma<8>(As, Bs, lane, wave, acc);
    }

    const int m = lane & 15, half = lane >> 4;
#pragma unroll
    for (int t = 0; t < 8; ++t) {
#pragma unroll
        for (int i = 0; i < 8; ++i) {
            const int row = mtile * 64 + wave * 16 + 8 * half + i;
            const int col = t * 16 + m;
            Og[(size_t)row * DD + col] = acc[t][i] + bv[col];
        }
    }
}

// ---------------------------------------------------------------------------
// Heads GEMM: flat[512][1920](bf16, stored [v][b][d]) x Wh[1920][64](bf16)
// scatter into the 4 concatenated outputs (6/32/10/3 cols)
// ---------------------------------------------------------------------------
__global__ void __launch_bounds__(128)
heads_kernel(const unsigned short* __restrict__ act,
             const unsigned short* __restrict__ Wh,
             const float* __restrict__ bh,
             float* __restrict__ out) {
    __shared__ unsigned short As[64 * LDSKA];
    __shared__ unsigned short Bs[64 * LDSKB];

    const int tid  = threadIdx.x;
    const int wave = tid >> 5;
    const int lane = tid & 31;
    const int mtile = blockIdx.x;

    v8f acc[4];
#pragma unroll
    for (int t = 0; t < 4; ++t) acc[t] = v8f{0.f,0.f,0.f,0.f,0.f,0.f,0.f,0.f};

    for (int ks = 0; ks < FF / 32; ++ks) {           // 60 K-steps
        const int v  = ks >> 2;
        const int db = (ks & 3) * 32;
        const unsigned short* Ag = act + (size_t)v * BB * DD + db;
        __syncthreads();
#pragma unroll
        for (int i = 0; i < 2; ++i) {                // A: 64 x 32 via async DMA
            const int chunk = tid + i * 128;
            const int row   = chunk >> 2;
            const int col   = (chunk & 3) * 8;
            async_ld_b128(&As[row * LDSKA + col],
                          Ag + (size_t)(mtile * 64 + row) * DD + col);
        }
#pragma unroll
        for (int i = 0; i < 2; ++i) {                // B: 32 x 64 transposed
            const int c  = tid + i * 128;
            const int kr = c >> 3;
            const int cb = (c & 7) * 8;
            const uint4 d = *(const uint4*)(Wh + (size_t)(ks * 32 + kr) * NHEAD + cb);
            const unsigned short* s = (const unsigned short*)&d;
#pragma unroll
            for (int j = 0; j < 8; ++j) Bs[(cb + j) * LDSKB + kr] = s[j];
        }
        async_wait0();
        __syncthreads();
        frag_mma<4>(As, Bs, lane, wave, acc);
    }

    const int m = lane & 15, half = lane >> 4;
#pragma unroll
    for (int t = 0; t < 4; ++t) {
#pragma unroll
        for (int i = 0; i < 8; ++i) {
            const int row = mtile * 64 + wave * 16 + 8 * half + i;
            const int col = t * 16 + m;
            const float val = acc[t][i] + bh[col];
            if (col < 6)        out[row * 6 + col] = val;
            else if (col < 38)  out[BB * 6 + row * 32 + (col - 6)] = val;
            else if (col < 48)  out[BB * 38 + row * 10 + (col - 38)] = val;
            else if (col < 51)  out[BB * 48 + row * 3 + (col - 48)] = val;
        }
    }
}

// ---------------------------------------------------------------------------
// Bilinear sample (align_corners, zero pad) + concat coords/score -> bf16 feat
// feat layout: [V][B][K0P]; grid = (V, B), block = 256
// ---------------------------------------------------------------------------
__global__ void sample_concat_kernel(const float* __restrict__ img,
                                     const float* __restrict__ coord,
                                     const float* __restrict__ score,
                                     unsigned short* __restrict__ feat) {
    const int v = blockIdx.x, b = blockIdx.y;
    const float px = coord[((size_t)b * VV + v) * 3 + 0];
    const float py = coord[((size_t)b * VV + v) * 3 + 1];
    const float x0 = floorf(px), y0 = floorf(py);
    const float x1 = x0 + 1.f,  y1 = y0 + 1.f;
    const float wx1 = px - x0, wx0 = 1.f - wx1;
    const float wy1 = py - y0, wy0 = 1.f - wy1;
    const float inx0 = (x0 >= 0.f && x0 <= WW - 1) ? 1.f : 0.f;
    const float inx1 = (x1 >= 0.f && x1 <= WW - 1) ? 1.f : 0.f;
    const float iny0 = (y0 >= 0.f && y0 <= HH - 1) ? 1.f : 0.f;
    const float iny1 = (y1 >= 0.f && y1 <= HH - 1) ? 1.f : 0.f;
    const int xi0 = min(max((int)x0, 0), WW - 1);
    const int xi1 = min(max((int)x1, 0), WW - 1);
    const int yi0 = min(max((int)y0, 0), HH - 1);
    const int yi1 = min(max((int)y1, 0), HH - 1);
    const float w00 = wy0 * wx0 * iny0 * inx0;
    const float w01 = wy0 * wx1 * iny0 * inx1;
    const float w10 = wy1 * wx0 * iny1 * inx0;
    const float w11 = wy1 * wx1 * iny1 * inx1;
    const float* base = img + (size_t)b * CC * (HH * WW);
    unsigned short* fo = feat + ((size_t)v * BB + b) * K0P;

    for (int c = threadIdx.x; c < K0P; c += 256) {
        float val;
        if (c < CC) {
            const float* p = base + (size_t)c * (HH * WW);
            val = p[yi0 * WW + xi0] * w00 + p[yi0 * WW + xi1] * w01 +
                  p[yi1 * WW + xi0] * w10 + p[yi1 * WW + xi1] * w11;
        } else if (c < CC + 3) {
            val = coord[((size_t)b * VV + v) * 3 + (c - CC)];
        } else if (c == CC + 3) {
            val = score[(size_t)b * VV + v];
        } else {
            val = 0.f;                               // K pad
        }
        fo[c] = f2bf(val);
    }
}

// ---------------------------------------------------------------------------
// BN batch statistics -> scale/shift; grid = V, block = 128 (one d per thread)
// ---------------------------------------------------------------------------
__global__ void bn_stats_kernel(const float* __restrict__ H,
                                const float* __restrict__ g,
                                const float* __restrict__ beta,
                                float* __restrict__ scale,
                                float* __restrict__ shift) {
    const int v = blockIdx.x, d = threadIdx.x;
    const float* p = H + (size_t)v * BB * DD + d;
    float s = 0.f, sq = 0.f;
    for (int b = 0; b < BB; ++b) { const float x = p[(size_t)b * DD]; s += x; sq += x * x; }
    const float mu  = s * (1.f / BB);
    const float var = sq * (1.f / BB) - mu * mu;
    const float sc  = g[v * DD + d] * rsqrtf(var + 1e-5f);
    scale[v * DD + d] = sc;
    shift[v * DD + d] = beta[v * DD + d] - mu * sc;
}

// ---------------------------------------------------------------------------
// BN apply + adjacency mix + ReLU (+ optional residual); grid = (B, V), block=128
// ---------------------------------------------------------------------------
__global__ void bn_adj_relu_kernel(const float* __restrict__ Hin,
                                   const float* __restrict__ scale,
                                   const float* __restrict__ shift,
                                   const float* __restrict__ adj,
                                   const float* __restrict__ Res,
                                   float* __restrict__ ActF,
                                   unsigned short* __restrict__ ActBF) {
    const int b = blockIdx.x, u = blockIdx.y, d = threadIdx.x;
    float accv = 0.f;
#pragma unroll
    for (int v = 0; v < VV; ++v) {
        const float x = Hin[((size_t)v * BB + b) * DD + d] * scale[v * DD + d] + shift[v * DD + d];
        accv += adj[u * VV + v] * x;
    }
    float r = fmaxf(accv, 0.f);
    const size_t o = ((size_t)u * BB + b) * DD + d;
    if (Res)  r += Res[o];
    if (ActF) ActF[o] = r;
    ActBF[o] = f2bf(r);
}

// ---------------------------------------------------------------------------
// Weight conversion / packing kernels
// ---------------------------------------------------------------------------
__global__ void w0_pad_kernel(const float* __restrict__ W0, unsigned short* __restrict__ out) {
    const long long total = (long long)VV * K0P * DD;
    const long long stride = (long long)gridDim.x * blockDim.x;
    for (long long i = (long long)blockIdx.x * blockDim.x + threadIdx.x; i < total; i += stride) {
        const int v = (int)(i / ((long long)K0P * DD));
        const int r = (int)(i % ((long long)K0P * DD));
        const int k = r / DD, n = r % DD;
        const float val = (k < DIN) ? W0[((size_t)v * DIN + k) * DD + n] : 0.f;
        out[i] = f2bf(val);
    }
}

__global__ void f32_to_bf16_kernel(const float* __restrict__ s,
                                   unsigned short* __restrict__ d, long long n) {
    const long long stride = (long long)gridDim.x * blockDim.x;
    for (long long i = (long long)blockIdx.x * blockDim.x + threadIdx.x; i < n; i += stride)
        d[i] = f2bf(s[i]);
}

__global__ void pack_heads_kernel(const float* __restrict__ Wr, const float* __restrict__ Wp,
                                  const float* __restrict__ Ws, const float* __restrict__ Wc,
                                  const float* __restrict__ br, const float* __restrict__ bp,
                                  const float* __restrict__ bs, const float* __restrict__ bc,
                                  unsigned short* __restrict__ Wh, float* __restrict__ bh) {
    const long long total = (long long)FF * NHEAD;
    const long long stride = (long long)gridDim.x * blockDim.x;
    for (long long i = (long long)blockIdx.x * blockDim.x + threadIdx.x; i < total; i += stride) {
        const int k = (int)(i >> 6), o = (int)(i & 63);
        float val = 0.f;
        if (o < 6)        val = Wr[k * 6 + o];
        else if (o < 38)  val = Wp[k * 32 + (o - 6)];
        else if (o < 48)  val = Ws[k * 10 + (o - 38)];
        else if (o < 51)  val = Wc[k * 3 + (o - 48)];
        Wh[i] = f2bf(val);
        if (i < NHEAD) {
            float bv = 0.f;
            const int oo = (int)i;
            if (oo < 6)       bv = br[oo];
            else if (oo < 38) bv = bp[oo - 6];
            else if (oo < 48) bv = bs[oo - 38];
            else if (oo < 51) bv = bc[oo - 48];
            bh[oo] = bv;
        }
    }
}

// ---------------------------------------------------------------------------
// Launch
// ---------------------------------------------------------------------------
extern "C" void kernel_launch(void* const* d_in, const int* in_sizes, int n_in,
                              void* d_out, int out_size, void* d_ws, size_t ws_size,
                              hipStream_t stream) {
    (void)in_sizes; (void)n_in; (void)out_size; (void)ws_size;
    const float* img    = (const float*)d_in[0];
    const float* coord  = (const float*)d_in[1];
    const float* score  = (const float*)d_in[2];
    const float* adj    = (const float*)d_in[3];
    const float* W0     = (const float*)d_in[4];
    const float* b0     = (const float*)d_in[5];
    const float* g0     = (const float*)d_in[6];
    const float* beta0  = (const float*)d_in[7];
    const float* Wres   = (const float*)d_in[8];
    const float* bres   = (const float*)d_in[9];
    const float* gres   = (const float*)d_in[10];
    const float* betar  = (const float*)d_in[11];
    const float* Wroot  = (const float*)d_in[12];
    const float* broot  = (const float*)d_in[13];
    const float* Wpose  = (const float*)d_in[14];
    const float* bpose  = (const float*)d_in[15];
    const float* Wshape = (const float*)d_in[16];
    const float* bshape = (const float*)d_in[17];
    const float* Wcam   = (const float*)d_in[18];
    const float* bcam   = (const float*)d_in[19];

    char* ws = (char*)d_ws;
    size_t off = 0;
    auto take = [&](size_t bytes) { size_t o = off; off = (off + bytes + 255) & ~(size_t)255; return o; };
    unsigned short* featbf  = (unsigned short*)(ws + take((size_t)VV * BB * K0P * 2));
    unsigned short* w0bf    = (unsigned short*)(ws + take((size_t)VV * K0P * DD * 2));
    unsigned short* wresbf  = (unsigned short*)(ws + take((size_t)8 * VV * DD * DD * 2));
    unsigned short* wheadbf = (unsigned short*)(ws + take((size_t)FF * NHEAD * 2));
    float*          headb   = (float*)(ws + take(NHEAD * 4));
    float*          gout    = (float*)(ws + take((size_t)VV * BB * DD * 4));
    float*          actF    = (float*)(ws + take((size_t)VV * BB * DD * 4));
    unsigned short* actBF   = (unsigned short*)(ws + take((size_t)VV * BB * DD * 2));
    float*          scale   = (float*)(ws + take((size_t)VV * DD * 4));
    float*          shift   = (float*)(ws + take((size_t)VV * DD * 4));

    // weight prep (bf16 conversion / padding / head packing)
    w0_pad_kernel<<<8192, 256, 0, stream>>>(W0, w0bf);
    f32_to_bf16_kernel<<<4096, 256, 0, stream>>>(Wres, wresbf, (long long)8 * VV * DD * DD);
    pack_heads_kernel<<<512, 256, 0, stream>>>(Wroot, Wpose, Wshape, Wcam,
                                               broot, bpose, bshape, bcam, wheadbf, headb);
    // sample + concat
    sample_concat_kernel<<<dim3(VV, BB), 256, 0, stream>>>(img, coord, score, featbf);

    // graph conv block 0 (DIN->D)
    gemm_bias_kernel<<<dim3(BB / 64, VV), 128, 0, stream>>>(featbf, w0bf, b0, gout, K0P);
    bn_stats_kernel<<<VV, DD, 0, stream>>>(gout, g0, beta0, scale, shift);
    bn_adj_relu_kernel<<<dim3(BB, VV), DD, 0, stream>>>(gout, scale, shift, adj,
                                                        nullptr, actF, actBF);
    // 4 residual blocks x 2 conv blocks
    for (int l = 0; l < 8; ++l) {
        const unsigned short* Wl = wresbf + (size_t)l * VV * DD * DD;
        const float* bl = bres + (size_t)l * VV * DD;
        const float* gl = gres + (size_t)l * VV * DD;
        const float* el = betar + (size_t)l * VV * DD;
        const bool second = (l & 1);
        gemm_bias_kernel<<<dim3(BB / 64, VV), 128, 0, stream>>>(actBF, Wl, bl, gout, DD);
        bn_stats_kernel<<<VV, DD, 0, stream>>>(gout, gl, el, scale, shift);
        bn_adj_relu_kernel<<<dim3(BB, VV), DD, 0, stream>>>(
            gout, scale, shift, adj,
            second ? actF : nullptr,          // residual added after 2nd conv
            second ? actF : nullptr,          // keep f32 residual intact after 1st conv
            actBF);
    }
    // regression heads
    heads_kernel<<<BB / 64, 128, 0, stream>>>(actBF, wheadbf, headb, (float*)d_out);
}
